// TPS_SpatialTransformerNetwork_31233002176871
// MI455X (gfx1250) — matrix operand order repaired and checked by reading, not compiled, not run on hardware
//
#include <hip/hip_runtime.h>

// ---------------------------------------------------------------------------
// TPS SpatialTransformerNetwork for MI455X (gfx1250, wave32, WMMA).
// Convs = implicit GEMM on v_wmma_f32_16x16x32_f16 (f32 accumulate).
// BN folded into weights; 2x2 maxpool fused into conv epilogue.
// Wave tile = 32(M) x 64(N): MT=2 M-tiles x NT=4 N-tiles -> B fragments
// amortized 2x, A fragments 4x. Convs 2-4 use channel-contiguous K-tiling:
// A fragment = 2x b128 loads, B fragment = contiguous 32B from pre-swizzled
// weights. Head (FC/TPS/grid-sample) is negligible FLOPs -> f32/f64 VALU.
// ---------------------------------------------------------------------------

typedef __attribute__((ext_vector_type(16))) _Float16 v16h;
typedef __attribute__((ext_vector_type(8)))  _Float16 v8h;
typedef __attribute__((ext_vector_type(8)))  float    v8f;

#define BN_EPS 1e-5f

__device__ __forceinline__ float reluf(float x) { return x > 0.f ? x : 0.f; }

// ---------------- conv1 weights: flat [K][COUT] f16, BN-folded -------------
__global__ void fold_weights_k(const float* __restrict__ w, const float* __restrict__ g,
                               const float* __restrict__ b, const float* __restrict__ m,
                               const float* __restrict__ v, _Float16* __restrict__ wh,
                               float* __restrict__ bias, int K, int COUT)
{
    int t = blockIdx.x * blockDim.x + threadIdx.x;
    if (t < COUT) {
        float s = g[t] * rsqrtf(v[t] + BN_EPS);
        bias[t] = b[t] - m[t] * s;
    }
    if (t < K * COUT) {
        int co = t % COUT;
        float s = g[co] * rsqrtf(v[co] + BN_EPS);
        wh[t] = (_Float16)(w[t] * s);
    }
}

// ------- convs 2-4: weights pre-swizzled into WMMA B-fragment order --------
// Layout: [kt][nt][lane][16 f16], kt over 9*CIN/32 K-tiles (K order = HWIO),
// lane: l15 = N within tile, khalf selects K half; element j -> K = khalf*8+(j&7)+(j>>3)*16.
// Each wave then fetches its B fragment as one contiguous 32-byte load.
__global__ void swizzle_weights_k(const float* __restrict__ w, const float* __restrict__ g,
                                  const float* __restrict__ b, const float* __restrict__ m,
                                  const float* __restrict__ v, _Float16* __restrict__ wswz,
                                  float* __restrict__ bias, int CIN, int COUT)
{
    int t = blockIdx.x * blockDim.x + threadIdx.x;
    if (t < COUT) {
        float s = g[t] * rsqrtf(v[t] + BN_EPS);
        bias[t] = b[t] - m[t] * s;
    }
    const int ntiles = COUT >> 4;
    const int total  = (9 * CIN / 32) * ntiles * 32; // one thread per (kt,nt,lane)
    if (t >= total) return;
    const int lane  = t & 31;
    const int nt    = (t >> 5) % ntiles;
    const int kt    = (t >> 5) / ntiles;
    const int l15   = lane & 15;
    const int khalf = lane >> 4;
    const int n     = (nt << 4) + l15;
    const float s   = g[n] * rsqrtf(v[n] + BN_EPS);
    _Float16* dst   = wswz + (size_t)t * 16;
    #pragma unroll
    for (int j = 0; j < 16; ++j) {
        const int kk = (kt << 5) + (khalf << 3) + (j & 7) + ((j >> 3) << 4); // global K (HWIO order)
        dst[j] = (_Float16)(w[kk * COUT + n] * s);
    }
}

// ---------------- conv3x3 SAME + BN + ReLU (+ fused 2x2 maxpool) -----------
// M = B*Ho*Wo, N = COUT, K = 9*CIN. One wave -> 32(M) x 64(N) tile (MT=2,NT=4).
template <int CIN, int COUT, int HI, int WI, bool POOL, bool INF32>
__global__ __launch_bounds__(256) void conv_wmma_k(
    const float* __restrict__ inF, const _Float16* __restrict__ inH,
    const _Float16* __restrict__ wh, const float* __restrict__ bias,
    _Float16* __restrict__ out)
{
    constexpr int K  = 9 * CIN;
    constexpr int HO = POOL ? HI / 2 : HI;
    constexpr int WO = POOL ? WI / 2 : WI;
    constexpr int NQ = POOL ? 4 : 1;
    constexpr int NTILES = COUT >> 4;

    const int lane  = threadIdx.x & 31;
    const int wave  = threadIdx.x >> 5;
    const int mt0   = (blockIdx.x * 8 + wave) * 2; // first of two M-tiles
    const int nt0   = blockIdx.y * 4;
    const int l15   = lane & 15;
    const int khalf = lane >> 4;

    // A-fragment rows owned by this lane, for both M-tiles
    int bb[2], yo[2], xo[2];
    #pragma unroll
    for (int mt = 0; mt < 2; ++mt) {
        const int mA = ((mt0 + mt) << 4) + l15;
        bb[mt] = mA / (HO * WO);
        const int rm = mA - bb[mt] * (HO * WO);
        yo[mt] = rm / WO;
        xo[mt] = rm - yo[mt] * WO;
    }

    float mx[2][4][8];

    #pragma unroll
    for (int q = 0; q < NQ; ++q) {
        v8f acc[2][4] = {};
        int y[2], x[2];
        #pragma unroll
        for (int mt = 0; mt < 2; ++mt) {
            y[mt] = POOL ? 2 * yo[mt] + (q >> 1) : yo[mt];
            x[mt] = POOL ? 2 * xo[mt] + (q & 1)  : xo[mt];
        }

        if constexpr (CIN >= 32) {
            // ---- channel-contiguous path: A = 2x contiguous 16B runs ------
            #pragma unroll
            for (int kpos = 0; kpos < 9; ++kpos) {
                const int kh = kpos / 3, kw = kpos - 3 * (kpos / 3);
                bool ok[2]; int pixbase[2];
                #pragma unroll
                for (int mt = 0; mt < 2; ++mt) {
                    const int yy = y[mt] + kh - 1, xx = x[mt] + kw - 1;
                    ok[mt] = (yy >= 0 && yy < HI && xx >= 0 && xx < WI);
                    pixbase[mt] = ((bb[mt] * HI + yy) * WI + xx) * CIN + (khalf << 3);
                }
                #pragma unroll
                for (int c0 = 0; c0 < CIN; c0 += 32) {
                    v16h af[2];
                    #pragma unroll
                    for (int mt = 0; mt < 2; ++mt) {
                        v8h lo = {}, hi = {};
                        if (ok[mt]) {
                            lo = *(const v8h*)(inH + pixbase[mt] + c0);
                            hi = *(const v8h*)(inH + pixbase[mt] + c0 + 16);
                        }
                        af[mt] = __builtin_shufflevector(
                            lo, hi, 0, 1, 2, 3, 4, 5, 6, 7, 8, 9, 10, 11, 12, 13, 14, 15);
                    }
                    const int kt = kpos * (CIN >> 5) + (c0 >> 5);
                    const _Float16* wbase =
                        wh + (((size_t)kt * NTILES + nt0) * 32 + lane) * 16;
                    #pragma unroll
                    for (int t = 0; t < 4; ++t) {
                        const v16h bf = *(const v16h*)(wbase + (size_t)t * 32 * 16);
                        #pragma unroll
                        for (int mt = 0; mt < 2; ++mt)
                            acc[mt][t] = __builtin_amdgcn_wmma_f32_16x16x32_f16(
                                false, af[mt], false, bf, (short)0, acc[mt][t], false, false);
                    }
                }
            }
        } else {
            // ---- generic masked-gather path (conv1: CIN=3, K=27) ----------
            for (int k0 = 0; k0 < K; k0 += 32) {
                v16h af[2];
                #pragma unroll
                for (int mt = 0; mt < 2; ++mt) {
                    #pragma unroll
                    for (int i = 0; i < 16; ++i) {
                        const int kk = k0 + (khalf << 3) + (i & 7) + ((i >> 3) << 4);
                        _Float16 hv = (_Float16)0.f;
                        if (kk < K) {
                            const int kh  = kk / (3 * CIN);
                            const int rem = kk - kh * 3 * CIN;
                            const int kw  = rem / CIN;
                            const int c   = rem - kw * CIN;
                            const int yy = y[mt] + kh - 1, xx = x[mt] + kw - 1;
                            if (yy >= 0 && yy < HI && xx >= 0 && xx < WI) {
                                const int idx = ((bb[mt] * HI + yy) * WI + xx) * CIN + c;
                                hv = INF32 ? (_Float16)inF[idx] : inH[idx];
                            }
                        }
                        af[mt][i] = hv;
                    }
                }
                #pragma unroll
                for (int t = 0; t < 4; ++t) {
                    const int n = ((nt0 + t) << 4) + l15;
                    v16h bf;
                    #pragma unroll
                    for (int j = 0; j < 16; ++j) {
                        const int kk = k0 + (khalf << 3) + (j & 7) + ((j >> 3) << 4);
                        bf[j] = (kk < K) ? wh[kk * COUT + n] : (_Float16)0.f;
                    }
                    #pragma unroll
                    for (int mt = 0; mt < 2; ++mt)
                        acc[mt][t] = __builtin_amdgcn_wmma_f32_16x16x32_f16(
                            false, af[mt], false, bf, (short)0, acc[mt][t], false, false);
                }
            }
        }

        #pragma unroll
        for (int mt = 0; mt < 2; ++mt)
            #pragma unroll
            for (int t = 0; t < 4; ++t)
                #pragma unroll
                for (int r = 0; r < 8; ++r)
                    mx[mt][t][r] = (q == 0) ? acc[mt][t][r]
                                            : fmaxf(mx[mt][t][r], acc[mt][t][r]);
    }

    // Epilogue: quadrant max == pool; bias + relu; f16 store.
    // C/D layout: VGPR r holds row M = r + 8*khalf, column N = lane&15.
    #pragma unroll
    for (int mt = 0; mt < 2; ++mt) {
        #pragma unroll
        for (int t = 0; t < 4; ++t) {
            const int n  = ((nt0 + t) << 4) + l15;
            const float bi = bias[n];
            #pragma unroll
            for (int r = 0; r < 8; ++r) {
                const int mo = ((mt0 + mt) << 4) + r + (khalf << 3);
                out[(size_t)mo * COUT + n] = (_Float16)reluf(mx[mt][t][r] + bi);
            }
        }
    }
}

// ---------------- global average pool over 4x12 -> (256,512) f32 ----------
__global__ void avgpool_k(const _Float16* __restrict__ a4, float* __restrict__ feat)
{
    int t = blockIdx.x * blockDim.x + threadIdx.x; // 256*512
    int b = t >> 9, c = t & 511;
    float s = 0.f;
    for (int p = 0; p < 48; ++p) s += (float)a4[((size_t)(b * 48 + p)) * 512 + c];
    feat[t] = s * (1.f / 48.f);
}

// ---------------- FC layers (tiny; f32 for control-point precision) --------
__global__ void fc1_k(const float* __restrict__ feat, const float* __restrict__ w,
                      const float* __restrict__ b, float* __restrict__ h)
{
    int t = blockIdx.x * blockDim.x + threadIdx.x; // 256*256
    int bi = t >> 8, j = t & 255;
    float s = b[j];
    for (int k = 0; k < 512; ++k) s += feat[bi * 512 + k] * w[k * 256 + j];
    h[t] = reluf(s);
}

__global__ void fc2_k(const float* __restrict__ h, const float* __restrict__ w,
                      const float* __restrict__ b, float* __restrict__ cp)
{
    int t = blockIdx.x * blockDim.x + threadIdx.x; // 256*40
    if (t >= 256 * 40) return;
    int bi = t / 40, j = t - bi * 40;
    float s = b[j];
    for (int k = 0; k < 256; ++k) s += h[bi * 256 + k] * w[k * 40 + j];
    cp[t] = s;
}

// ---------------- TPS constants (not inputs -> computed on device) --------
__device__ __forceinline__ void ctrl_pt(int i, double* cx, double* cy)
{
    if (i < 10) { *cx = -1.0 + 2.0 * i / 9.0; *cy = -(double)i / 9.0; }
    else        { int j = i - 10; *cx = -1.0 + 2.0 * j / 9.0; *cy = 1.0 - (double)j / 9.0; }
}

// 23x23 inverse of the TPS system matrix, double Gauss-Jordan w/ pivoting.
__global__ void tps_inv_k(float* __restrict__ invOut)
{
    if (threadIdx.x != 0 || blockIdx.x != 0) return;
    double A[23][46];
    double Cx[20], Cy[20];
    for (int i = 0; i < 20; ++i) ctrl_pt(i, &Cx[i], &Cy[i]);
    for (int i = 0; i < 23; ++i)
        for (int j = 0; j < 46; ++j) A[i][j] = 0.0;
    for (int i = 0; i < 20; ++i) {
        A[i][0] = 1.0; A[i][1] = Cx[i]; A[i][2] = Cy[i];
        for (int j = 0; j < 20; ++j) {
            double dx = Cx[i] - Cx[j], dy = Cy[i] - Cy[j];
            double d2 = dx * dx + dy * dy;
            A[i][3 + j] = (i == j) ? 0.0 : d2 * log(d2);
        }
    }
    for (int j = 0; j < 20; ++j) {
        A[20][3 + j] = Cx[j]; A[21][3 + j] = Cy[j]; A[22][3 + j] = 1.0;
    }
    for (int i = 0; i < 23; ++i) A[i][23 + i] = 1.0;
    for (int col = 0; col < 23; ++col) {
        int piv = col; double best = fabs(A[col][col]);
        for (int r = col + 1; r < 23; ++r) {
            double a = fabs(A[r][col]);
            if (a > best) { best = a; piv = r; }
        }
        if (piv != col)
            for (int j = 0; j < 46; ++j) { double tm = A[col][j]; A[col][j] = A[piv][j]; A[piv][j] = tm; }
        double iv = 1.0 / A[col][col];
        for (int j = 0; j < 46; ++j) A[col][j] *= iv;
        for (int r = 0; r < 23; ++r) {
            if (r == col) continue;
            double f = A[r][col];
            if (f != 0.0)
                for (int j = 0; j < 46; ++j) A[r][j] -= f * A[col][j];
        }
    }
    for (int i = 0; i < 23; ++i)
        for (int j = 0; j < 23; ++j) invOut[i * 23 + j] = (float)A[i][23 + j];
}

// P_hat (3200 x 23): [1, x, y, 2*d^2*log(d + 1e-6) per control point]
__global__ void phat_k(float* __restrict__ ph)
{
    int n = blockIdx.x * blockDim.x + threadIdx.x;
    if (n >= 3200) return;
    int ny = n / 100, nx = n - ny * 100;
    double x = (2.0 * nx - 99.0) / 100.0;
    double y = (2.0 * ny - 31.0) / 32.0;
    ph[n * 23 + 0] = 1.f;
    ph[n * 23 + 1] = (float)x;
    ph[n * 23 + 2] = (float)y;
    for (int f = 0; f < 20; ++f) {
        double cx, cy; ctrl_pt(f, &cx, &cy);
        double dx = x - cx, dy = y - cy;
        double d2 = dx * dx + dy * dy;
        ph[n * 23 + 3 + f] = (float)(2.0 * d2 * log(sqrt(d2) + 1e-6));
    }
}

// T = inv_delta_C @ [C'; 0]  -> (256, 23, 2)
__global__ void tmat_k(const float* __restrict__ inv, const float* __restrict__ cp,
                       float* __restrict__ T)
{
    int t = blockIdx.x * blockDim.x + threadIdx.x; // 256*46
    if (t >= 256 * 46) return;
    int b = t / 46, rem = t - b * 46;
    int i = rem >> 1, k = rem & 1;
    float s = 0.f;
    for (int j = 0; j < 20; ++j) s += inv[i * 23 + j] * cp[b * 40 + j * 2 + k];
    T[t] = s; // layout matches ((b*23+i)*2+k)
}

// grid = P_hat @ T, then bilinear border-mode sample of the input image.
__global__ void sample_k(const float* __restrict__ img, const float* __restrict__ ph,
                         const float* __restrict__ T, float* __restrict__ out)
{
    int t = blockIdx.x * blockDim.x + threadIdx.x; // 256*3200
    if (t >= 256 * 3200) return;
    int b = t / 3200, n = t - b * 3200;
    const float* phr = ph + n * 23;
    const float* Tb  = T + b * 46;
    float px = 0.f, py = 0.f;
    for (int j = 0; j < 23; ++j) {
        float p = phr[j];
        px += p * Tb[j * 2 + 0];
        py += p * Tb[j * 2 + 1];
    }
    float gx = (px + 1.f) * 50.f - 0.5f;
    float gy = (py + 1.f) * 16.f - 0.5f;
    float fx0 = floorf(gx), fy0 = floorf(gy);
    float wx = gx - fx0, wy = gy - fy0;
    int x0 = (int)fx0, y0 = (int)fy0;
    int x0i = min(max(x0, 0), 99);
    int x1i = min(max(x0 + 1, 0), 99);
    int y0i = min(max(y0, 0), 31);
    int y1i = min(max(y0 + 1, 0), 31);
    const float* ib = img + (size_t)b * 32 * 100 * 3;
    #pragma unroll
    for (int c = 0; c < 3; ++c) {
        float v00 = ib[(y0i * 100 + x0i) * 3 + c];
        float v01 = ib[(y0i * 100 + x1i) * 3 + c];
        float v10 = ib[(y1i * 100 + x0i) * 3 + c];
        float v11 = ib[(y1i * 100 + x1i) * 3 + c];
        float top = v00 * (1.f - wx) + v01 * wx;
        float bot = v10 * (1.f - wx) + v11 * wx;
        out[(size_t)t * 3 + c] = top * (1.f - wy) + bot * wy;
    }
}

// ---------------------------------------------------------------------------
extern "C" void kernel_launch(void* const* d_in, const int* in_sizes, int n_in,
                              void* d_out, int out_size, void* d_ws, size_t ws_size,
                              hipStream_t stream)
{
    const float* img  = (const float*)d_in[0];
    const float* c1w = (const float*)d_in[1];
    const float* c1g = (const float*)d_in[2];
    const float* c1b = (const float*)d_in[3];
    const float* c1m = (const float*)d_in[4];
    const float* c1v = (const float*)d_in[5];
    const float* c2w = (const float*)d_in[6];
    const float* c2g = (const float*)d_in[7];
    const float* c2b = (const float*)d_in[8];
    const float* c2m = (const float*)d_in[9];
    const float* c2v = (const float*)d_in[10];
    const float* c3w = (const float*)d_in[11];
    const float* c3g = (const float*)d_in[12];
    const float* c3b = (const float*)d_in[13];
    const float* c3m = (const float*)d_in[14];
    const float* c3v = (const float*)d_in[15];
    const float* c4w = (const float*)d_in[16];
    const float* c4g = (const float*)d_in[17];
    const float* c4b = (const float*)d_in[18];
    const float* c4m = (const float*)d_in[19];
    const float* c4v = (const float*)d_in[20];
    const float* f1w = (const float*)d_in[21];
    const float* f1b = (const float*)d_in[22];
    const float* f2w = (const float*)d_in[23];
    const float* f2b = (const float*)d_in[24];

    // ---- workspace layout (~63 MB; pools fused into convs, no pre-pool bufs)
    char* ws = (char*)d_ws;
    size_t off = 0;
    auto alloc = [&](size_t bytes) -> char* {
        char* p = ws + off;
        off += (bytes + 255) & ~(size_t)255;
        return p;
    };
    _Float16* p1  = (_Float16*)alloc(sizeof(_Float16) * (size_t)256 * 16 * 50 * 64);
    _Float16* p2  = (_Float16*)alloc(sizeof(_Float16) * (size_t)256 * 8 * 25 * 128);
    _Float16* p3  = (_Float16*)alloc(sizeof(_Float16) * (size_t)256 * 4 * 12 * 256);
    _Float16* a4  = (_Float16*)alloc(sizeof(_Float16) * (size_t)256 * 4 * 12 * 512);
    _Float16* wh1 = (_Float16*)alloc(sizeof(_Float16) * 27 * 64);
    _Float16* wh2 = (_Float16*)alloc(sizeof(_Float16) * 576 * 128);   // swizzled
    _Float16* wh3 = (_Float16*)alloc(sizeof(_Float16) * 1152 * 256);  // swizzled
    _Float16* wh4 = (_Float16*)alloc(sizeof(_Float16) * 2304 * 512);  // swizzled
    float* bs1  = (float*)alloc(sizeof(float) * 64);
    float* bs2  = (float*)alloc(sizeof(float) * 128);
    float* bs3  = (float*)alloc(sizeof(float) * 256);
    float* bs4  = (float*)alloc(sizeof(float) * 512);
    float* feat = (float*)alloc(sizeof(float) * 256 * 512);
    float* h1   = (float*)alloc(sizeof(float) * 256 * 256);
    float* cp   = (float*)alloc(sizeof(float) * 256 * 40);
    float* invm = (float*)alloc(sizeof(float) * 23 * 23);
    float* ph   = (float*)alloc(sizeof(float) * 3200 * 23);
    float* Tm   = (float*)alloc(sizeof(float) * 256 * 23 * 2);

    // ---- constants / weight prep
    fold_weights_k<<<(27 * 64 + 255) / 256, 256, 0, stream>>>(c1w, c1g, c1b, c1m, c1v, wh1, bs1, 27, 64);
    swizzle_weights_k<<<(576 * 128 / 16 + 255) / 256, 256, 0, stream>>>(c2w, c2g, c2b, c2m, c2v, wh2, bs2, 64, 128);
    swizzle_weights_k<<<(1152 * 256 / 16 + 255) / 256, 256, 0, stream>>>(c3w, c3g, c3b, c3m, c3v, wh3, bs3, 128, 256);
    swizzle_weights_k<<<(2304 * 512 / 16 + 255) / 256, 256, 0, stream>>>(c4w, c4g, c4b, c4m, c4v, wh4, bs4, 256, 512);
    tps_inv_k<<<1, 1, 0, stream>>>(invm);
    phat_k<<<13, 256, 0, stream>>>(ph);

    // ---- localization CNN (WMMA implicit GEMM, pool fused)
    // blocks.x = Mtiles/16 (8 waves x MT=2 per block); blocks.y = COUT/64
    conv_wmma_k<3, 64, 32, 100, true, true><<<dim3(800, 1), 256, 0, stream>>>(img, nullptr, wh1, bs1, p1);
    conv_wmma_k<64, 128, 16, 50, true, false><<<dim3(200, 2), 256, 0, stream>>>(nullptr, p1, wh2, bs2, p2);
    conv_wmma_k<128, 256, 8, 25, true, false><<<dim3(48, 4), 256, 0, stream>>>(nullptr, p2, wh3, bs3, p3);
    conv_wmma_k<256, 512, 4, 12, false, false><<<dim3(48, 8), 256, 0, stream>>>(nullptr, p3, wh4, bs4, a4);

    // ---- head + TPS + sampling
    avgpool_k<<<512, 256, 0, stream>>>(a4, feat);
    fc1_k<<<256, 256, 0, stream>>>(feat, f1w, f1b, h1);
    fc2_k<<<40, 256, 0, stream>>>(h1, f2w, f2b, cp);
    tmat_k<<<46, 256, 0, stream>>>(invm, cp, Tm);
    sample_k<<<3200, 256, 0, stream>>>(img, ph, Tm, (float*)d_out);

    (void)in_sizes; (void)n_in; (void)out_size; (void)ws_size;
}
